// MambaBlock_49881750176423
// MI455X (gfx1250) — matrix operand order repaired and checked by reading, not compiled, not run on hardware
//
#include <hip/hip_runtime.h>
#include <hip/hip_bf16.h>
#include <stdint.h>

typedef __bf16 bf16_t;
typedef __attribute__((ext_vector_type(16))) __bf16 v16bf;
typedef __attribute__((ext_vector_type(8)))  __bf16 v8bf;
typedef __attribute__((ext_vector_type(8)))  float   v8f;

#define TILE_M 128
#define TILE_N 128
#define TILE_K 64
#define LDSK   (TILE_K + 8)   // 72 bf16 = 144 B row stride (16B aligned)

// ---------------------------------------------------------------- casts
__global__ __launch_bounds__(256)
void k_cast_bf16(const float* __restrict__ in, bf16_t* __restrict__ out, long n) {
  long i = (long)blockIdx.x * 256 + threadIdx.x;
  if (i < n) out[i] = (bf16_t)in[i];
}

// in: R x C fp32 row-major  ->  out: C x R bf16 row-major
__global__ __launch_bounds__(256)
void k_transpose_cast(const float* __restrict__ in, bf16_t* __restrict__ out,
                      int R, int C) {
  __shared__ float tile[32][33];
  int c0 = blockIdx.x * 32, r0 = blockIdx.y * 32;
  int x = threadIdx.x;
  for (int i = threadIdx.y; i < 32; i += 8) {
    int r = r0 + i, c = c0 + x;
    tile[i][x] = (r < R && c < C) ? in[(size_t)r * C + c] : 0.f;
  }
  __syncthreads();
  for (int i = threadIdx.y; i < 32; i += 8) {
    int c = c0 + i, r = r0 + x;
    if (c < C && r < R) out[(size_t)c * R + r] = (bf16_t)tile[x][i];
  }
}

// ---------------------------------------------------------------- GEMM
// C[M,N] (f32, row-major) = A[M,K] (bf16, row-major) * Bt[N,K]^T (bf16, N-major)
// M multiple of 128, K multiple of 64. N guarded.
// Double-buffered K tiles staged with GLOBAL_LOAD_ASYNC_TO_LDS_B128 (ASYNCcnt).
__global__ __launch_bounds__(256)
void k_gemm_bf16f32(const bf16_t* __restrict__ A, const bf16_t* __restrict__ Bt,
                    float* __restrict__ C, int M, int N, int K) {
  __shared__ __align__(16) bf16_t As[2 * TILE_M * LDSK];
  __shared__ __align__(16) bf16_t Bs[2 * TILE_N * LDSK];
  __shared__ __align__(16) bf16_t Dump[32];   // sink for OOB async loads

  const int tid  = threadIdx.x;
  const int lane = tid & 31;
  const int wave = tid >> 5;
  const int hl   = lane >> 4;        // lane half (0/1)
  const int r16  = lane & 15;
  const int waveM = wave & 1;        // 2 waves along M (each 64 rows)
  const int waveN = wave >> 1;       // 4 waves along N (each 32 cols)
  const int m0 = blockIdx.y * TILE_M;
  const int n0 = blockIdx.x * TILE_N;

  // per-thread staging slice: row = tid/2, 64 contiguous bytes at col (tid&1)*32
  const int ldRow = tid >> 1;
  const int ldCol = (tid & 1) * 32;
  const bool bValid = (n0 + ldRow) < N;
  const uint32_t dumpAddr = (uint32_t)(uintptr_t)Dump;

  // zero-fill OOB B rows once (both buffers); async loads never touch them
  if (!bValid) {
#pragma unroll
    for (int i = 0; i < 4; ++i) {
      v8bf z = {};
      *(v8bf*)(Bs + ldRow * LDSK + ldCol + i * 8) = z;
      *(v8bf*)(Bs + TILE_N * LDSK + ldRow * LDSK + ldCol + i * 8) = z;
    }
  }

  auto issue_tile = [&](int k0, int buf) {
    const bf16_t* ga = A + (size_t)(m0 + ldRow) * K + k0 + ldCol;
    bf16_t* la = As + buf * (TILE_M * LDSK) + ldRow * LDSK + ldCol;
#pragma unroll
    for (int i = 0; i < 4; ++i) {
      asm volatile("global_load_async_to_lds_b128 %0, %1, off"
                   :: "v"((uint32_t)(uintptr_t)(la + i * 8)), "v"(ga + i * 8)
                   : "memory");
    }
    const bf16_t* gb = bValid ? (Bt + (size_t)(n0 + ldRow) * K + k0 + ldCol) : Bt;
    bf16_t* lb = Bs + buf * (TILE_N * LDSK) + ldRow * LDSK + ldCol;
#pragma unroll
    for (int i = 0; i < 4; ++i) {
      uint32_t dst = bValid ? (uint32_t)(uintptr_t)(lb + i * 8) : dumpAddr;
      const bf16_t* src = bValid ? (gb + i * 8) : gb;
      asm volatile("global_load_async_to_lds_b128 %0, %1, off"
                   :: "v"(dst), "v"(src)
                   : "memory");
    }
  };

  v8f acc[4][2];
#pragma unroll
  for (int tm = 0; tm < 4; ++tm)
#pragma unroll
    for (int tn = 0; tn < 2; ++tn) { v8f z = {}; acc[tm][tn] = z; }

  issue_tile(0, 0);
  int cur = 0;
  for (int k0 = 0; k0 < K; k0 += TILE_K) {
    // my loads for the current tile are complete ...
    asm volatile("s_wait_asynccnt 0x0" ::: "memory");
    // ... and after the barrier everyone's are, and everyone finished the
    // previous compute, so buf cur^1 may be overwritten.
    __syncthreads();
    if (k0 + TILE_K < K) issue_tile(k0 + TILE_K, cur ^ 1);

    const bf16_t* Asb = As + cur * (TILE_M * LDSK);
    const bf16_t* Bsb = Bs + cur * (TILE_N * LDSK);

#pragma unroll
    for (int kk = 0; kk < TILE_K; kk += 32) {
      // A fragments: lane(row=r16) gets K = kk + {hl*8..hl*8+7, 16+hl*8..}
      v16bf af[4];
#pragma unroll
      for (int tm = 0; tm < 4; ++tm) {
        const bf16_t* p = Asb + (waveM * 64 + tm * 16 + r16) * LDSK + kk;
        v8bf lo = *(const v8bf*)(p + hl * 8);
        v8bf hi = *(const v8bf*)(p + 16 + hl * 8);
#pragma unroll
        for (int e = 0; e < 8; ++e) { af[tm][e] = lo[e]; af[tm][8 + e] = hi[e]; }
      }
      // B fragments: lane(col=r16) gets K = kk + {hl*8..hl*8+7, 16+hl*8..}
      v16bf bfv[2];
#pragma unroll
      for (int tn = 0; tn < 2; ++tn) {
        const bf16_t* p = Bsb + (waveN * 32 + tn * 16 + r16) * LDSK + kk;
        v8bf lo = *(const v8bf*)(p + hl * 8);
        v8bf hi = *(const v8bf*)(p + 16 + hl * 8);
#pragma unroll
        for (int e = 0; e < 8; ++e) { bfv[tn][e] = lo[e]; bfv[tn][8 + e] = hi[e]; }
      }
#pragma unroll
      for (int tm = 0; tm < 4; ++tm)
#pragma unroll
        for (int tn = 0; tn < 2; ++tn)
          acc[tm][tn] = __builtin_amdgcn_wmma_f32_16x16x32_bf16(
              false, af[tm], false, bfv[tn], (short)0, acc[tm][tn], false, false);
    }
    cur ^= 1;
  }

  // C layout: VGPR r -> M = hl*8 + r, N = r16
#pragma unroll
  for (int tm = 0; tm < 4; ++tm)
#pragma unroll
    for (int tn = 0; tn < 2; ++tn) {
      int col = n0 + waveN * 32 + tn * 16 + r16;
      if (col < N) {
        int rowb = m0 + waveM * 64 + tm * 16 + hl * 8;
#pragma unroll
        for (int r = 0; r < 8; ++r)
          C[(size_t)(rowb + r) * N + col] = acc[tm][tn][r];
      }
    }
}

// ---------------------------------------------------------------- conv + SiLU
__global__ __launch_bounds__(256)
void k_conv_silu(const float* __restrict__ xz, const float* __restrict__ cw,
                 const float* __restrict__ cb, float* __restrict__ xcf,
                 bf16_t* __restrict__ xcb) {
  const int d  = blockIdx.x * 256 + threadIdx.x;   // 0..2047
  const int b  = blockIdx.y;                       // 0..3
  const int l0 = blockIdx.z * 256;
  const float w0 = cw[d * 4 + 0], w1 = cw[d * 4 + 1];
  const float w2 = cw[d * 4 + 2], w3 = cw[d * 4 + 3];
  const float bias = cb[d];
  const size_t rowb = (size_t)b * 2048;            // b*L

  float xm3 = (l0 - 3 >= 0) ? xz[(rowb + l0 - 3) * 4096 + d] : 0.f;
  float xm2 = (l0 - 2 >= 0) ? xz[(rowb + l0 - 2) * 4096 + d] : 0.f;
  float xm1 = (l0 - 1 >= 0) ? xz[(rowb + l0 - 1) * 4096 + d] : 0.f;
  for (int l = l0; l < l0 + 256; ++l) {
    float x0 = xz[(rowb + l) * 4096 + d];
    float v  = w0 * xm3 + w1 * xm2 + w2 * xm1 + w3 * x0 + bias;
    float s  = v / (1.f + __expf(-v));             // silu
    size_t o = (rowb + l) * 2048 + d;
    xcf[o] = s;
    xcb[o] = (bf16_t)s;
    xm3 = xm2; xm2 = xm1; xm1 = x0;
  }
}

// ---------------------------------------------------------------- selective scan
__global__ __launch_bounds__(256)
void k_scan(const float* __restrict__ dtpre, const float* __restrict__ xconv,
            const float* __restrict__ bc, const float* __restrict__ xz,
            const float* __restrict__ bdt, const float* __restrict__ A_log,
            const float* __restrict__ Dp, bf16_t* __restrict__ yout) {
  const int b = blockIdx.y;
  const int d = blockIdx.x * 256 + threadIdx.x;
  __shared__ float bcs[32][32];

  float An[16];
#pragma unroll
  for (int n = 0; n < 16; ++n) An[n] = -__expf(A_log[n]);
  const float Dd  = Dp[d];
  const float bdd = bdt[d];
  float h[16];
#pragma unroll
  for (int n = 0; n < 16; ++n) h[n] = 0.f;

  const size_t rowb = (size_t)b * 2048;            // b*L
  for (int c = 0; c < 64; ++c) {                   // L / 32 chunks
    const int l0 = c * 32;
    __syncthreads();
    for (int i = threadIdx.x; i < 32 * 32; i += 256)
      bcs[i >> 5][i & 31] = bc[((rowb + l0 + (i >> 5)) << 5) + (i & 31)];
    __syncthreads();
#pragma unroll 4
    for (int l = 0; l < 32; ++l) {
      size_t idx = (rowb + l0 + l) * 2048 + d;
      float dtp = dtpre[idx] + bdd;
      float u   = xconv[idx];
      float zv  = xz[(rowb + l0 + l) * 4096 + 2048 + d];
      float dt  = (dtp > 20.f) ? dtp : log1pf(__expf(dtp));   // softplus
      float du  = dt * u;
      float y   = 0.f;
#pragma unroll
      for (int n = 0; n < 16; ++n) {
        float dA = __expf(dt * An[n]);
        h[n] = dA * h[n] + du * bcs[l][n];
        y += h[n] * bcs[l][16 + n];
      }
      y += u * Dd;
      float sg = zv / (1.f + __expf(-zv));
      yout[idx] = (bf16_t)(y * sg);
    }
  }
}

// ---------------------------------------------------------------- launcher
extern "C" void kernel_launch(void* const* d_in, const int* in_sizes, int n_in,
                              void* d_out, int out_size, void* d_ws, size_t ws_size,
                              hipStream_t stream) {
  (void)in_sizes; (void)n_in; (void)out_size; (void)ws_size;
  const float* x      = (const float*)d_in[0];
  const float* W_in   = (const float*)d_in[1];
  const float* conv_w = (const float*)d_in[2];
  const float* conv_b = (const float*)d_in[3];
  const float* W_x    = (const float*)d_in[4];
  const float* W_dt   = (const float*)d_in[5];
  const float* b_dt   = (const float*)d_in[6];
  const float* A_log  = (const float*)d_in[7];
  const float* Dv     = (const float*)d_in[8];
  const float* W_out  = (const float*)d_in[9];
  float* out = (float*)d_out;

  const long MR = 8192;          // B * L
  const int  DM = 1024, DI = 2048;

  char* base = (char*)d_ws;
  size_t off = 0;
  auto alloc = [&](size_t bytes) -> char* {
    char* p = base + off;
    off = (off + bytes + 255) & ~(size_t)255;
    return p;
  };
  bf16_t* xb    = (bf16_t*)alloc((size_t)MR * DM * 2);
  bf16_t* wint  = (bf16_t*)alloc((size_t)2 * DI * DM * 2);
  bf16_t* wdtt  = (bf16_t*)alloc((size_t)DI * DI * 2);
  bf16_t* wxt   = (bf16_t*)alloc((size_t)32 * DI * 2);
  bf16_t* woutt = (bf16_t*)alloc((size_t)DM * DI * 2);
  float*  xz    = (float*)alloc((size_t)MR * 2 * DI * 4);
  float*  xcf   = (float*)alloc((size_t)MR * DI * 4);
  bf16_t* xcb   = (bf16_t*)alloc((size_t)MR * DI * 2);
  float*  dtp   = (float*)alloc((size_t)MR * DI * 4);
  float*  bcb   = (float*)alloc((size_t)MR * 32 * 4);
  bf16_t* yb    = (bf16_t*)alloc((size_t)MR * DI * 2);

  dim3 tb(32, 8);
  // precision conversion + weight pre-transpose
  k_cast_bf16<<<dim3((unsigned)((MR * DM) / 256)), 256, 0, stream>>>(x, xb, MR * DM);
  k_transpose_cast<<<dim3((2 * DI) / 32, DM / 32), tb, 0, stream>>>(W_in, wint, DM, 2 * DI);
  k_transpose_cast<<<dim3(DI / 32, DI / 32), tb, 0, stream>>>(W_dt, wdtt, DI, DI);
  k_transpose_cast<<<dim3(1, DI / 32), tb, 0, stream>>>(W_x, wxt, DI, 32);
  k_transpose_cast<<<dim3(DM / 32, DI / 32), tb, 0, stream>>>(W_out, woutt, DI, DM);

  // xz = x @ W_in            (8192 x 4096, K=1024)
  k_gemm_bf16f32<<<dim3((2 * DI) / TILE_N, MR / TILE_M), 256, 0, stream>>>(
      xb, wint, xz, (int)MR, 2 * DI, DM);
  // causal depthwise conv + SiLU
  k_conv_silu<<<dim3(DI / 256, 4, 2048 / 256), 256, 0, stream>>>(
      xz, conv_w, conv_b, xcf, xcb);
  // dt_pre = x_conv @ W_dt   (8192 x 2048, K=2048)
  k_gemm_bf16f32<<<dim3(DI / TILE_N, MR / TILE_M), 256, 0, stream>>>(
      xcb, wdtt, dtp, (int)MR, DI, DI);
  // BC = x_conv @ W_x        (8192 x 32, K=2048)
  k_gemm_bf16f32<<<dim3(1, MR / TILE_M), 256, 0, stream>>>(
      xcb, wxt, bcb, (int)MR, 32, DI);
  // selective scan + gating
  k_scan<<<dim3(DI / 256, 4), 256, 0, stream>>>(
      dtp, xcf, bcb, xz, b_dt, A_log, Dv, yb);
  // out = y @ W_out          (8192 x 1024, K=2048)
  k_gemm_bf16f32<<<dim3(DM / TILE_N, MR / TILE_M), 256, 0, stream>>>(
      yb, woutt, out, (int)MR, DM, DI);
}
